// OptimizedDSMoE_57818849738788
// MI455X (gfx1250) — matrix operand order repaired and checked by reading, not compiled, not run on hardware
//
#include <hip/hip_runtime.h>
#include <hip/hip_bf16.h>
#include <math.h>

// ---------------- problem constants ----------------
#define T_TOK   8192        // BATCH*SEQ
#define DM      1024        // d_model
#define DFF     4096        // d_ff
#define NE      8           // experts
#define TOPK    2
#define CAP     16896       // 16384 pairs + 8*64 alignment padding (264 * 64)
#define NROWT   264         // CAP/64 row tiles

typedef __attribute__((ext_vector_type(16))) __bf16 v16bf;
typedef __attribute__((ext_vector_type(2)))  __bf16 v2bf;
typedef __attribute__((ext_vector_type(8)))  float  v8f;
typedef __attribute__((ext_vector_type(4)))  int    v4i;
typedef __attribute__((address_space(1))) v4i* gv4i_p;   // global int4*
typedef __attribute__((address_space(3))) v4i* sv4i_p;   // LDS int4*

#if __has_builtin(__builtin_amdgcn_global_load_async_to_lds_b128)
#define USE_ASYNC_LDS 1
#else
#define USE_ASYNC_LDS 0
#endif

// ---------------- workspace layout (bytes) ----------------
#define OFF_PROBS   ((size_t)0x0)         // T*8 f32          = 256 KB
#define OFF_SELE    ((size_t)0x40000)     // T*2 i32          = 64 KB
#define OFF_SELW    ((size_t)0x50000)     // T*2 f32          = 64 KB
#define OFF_CNT     ((size_t)0x60000)     // 8 i32
#define OFF_OFFS    ((size_t)0x60100)     // 9 i32
#define OFF_PTOK    ((size_t)0x60200)     // CAP i32          = 66 KB
#define OFF_TSLOT   ((size_t)0x78000)     // T*2 i32          = 64 KB
#define OFF_XG      ((size_t)0x100000)    // CAP*1024 bf16    = 34.6 MB
#define OFF_H       ((size_t)0x2300000)   // CAP*4096 bf16    = 138.4 MB
#define OFF_Y       ((size_t)0xA800000)   // CAP*1024 f32     = 69.2 MB
// total ~ 235 MB

__device__ __forceinline__ unsigned short bf16_of(float f) {
    return __builtin_bit_cast(unsigned short, (__bf16)f);   // native cvt (RNE)
}
__device__ __forceinline__ unsigned pk2(float a, float b) {
    v2bf v; v[0] = (__bf16)a; v[1] = (__bf16)b;
    return __builtin_bit_cast(unsigned, v);
}
__device__ __forceinline__ v16bf mk16(uint4 lo, uint4 hi) {
    union { uint4 q[2]; v16bf v; } u; u.q[0] = lo; u.q[1] = hi; return u.v;
}
__device__ __forceinline__ void wait_async0() {
#if USE_ASYNC_LDS
#if __has_builtin(__builtin_amdgcn_s_wait_asynccnt)
    __builtin_amdgcn_s_wait_asynccnt(0);
#else
    asm volatile("s_wait_asynccnt 0x0" ::: "memory");
#endif
#endif
}

// ================= 1) gate: wave-per-token GEMV + softmax + top2 ===========
__global__ void moe_gate(const float* __restrict__ x, const float* __restrict__ wg,
                         float* __restrict__ probs, int* __restrict__ sel_e,
                         float* __restrict__ sel_w) {
    int gw   = (blockIdx.x * blockDim.x + threadIdx.x) >> 5;   // wave id = token
    int lane = threadIdx.x & 31;
    if (gw >= T_TOK) return;
    const float* xr = x + (size_t)gw * DM;

    float acc[NE];
#pragma unroll
    for (int e = 0; e < NE; ++e) acc[e] = 0.f;
    for (int i = lane; i < DM; i += 32) {
        float xv = xr[i];
        float4 w0 = *(const float4*)(wg + (size_t)i * NE);
        float4 w1 = *(const float4*)(wg + (size_t)i * NE + 4);
        acc[0] += xv * w0.x; acc[1] += xv * w0.y; acc[2] += xv * w0.z; acc[3] += xv * w0.w;
        acc[4] += xv * w1.x; acc[5] += xv * w1.y; acc[6] += xv * w1.z; acc[7] += xv * w1.w;
    }
#pragma unroll
    for (int off = 16; off > 0; off >>= 1)
#pragma unroll
        for (int e = 0; e < NE; ++e) acc[e] += __shfl_xor(acc[e], off, 32);

    float mx = acc[0];
#pragma unroll
    for (int e = 1; e < NE; ++e) mx = fmaxf(mx, acc[e]);
    float p[NE], s = 0.f;
#pragma unroll
    for (int e = 0; e < NE; ++e) { p[e] = expf(acc[e] - mx); s += p[e]; }
    float inv = 1.f / s;
#pragma unroll
    for (int e = 0; e < NE; ++e) p[e] *= inv;

    int i0 = 0; float v0 = p[0];
#pragma unroll
    for (int e = 1; e < NE; ++e) if (p[e] > v0) { v0 = p[e]; i0 = e; }
    int i1 = -1; float v1 = -1.f;
#pragma unroll
    for (int e = 0; e < NE; ++e) if (e != i0 && p[e] > v1) { v1 = p[e]; i1 = e; }
    float z = 1.f / (v0 + v1 + 1e-8f);

    if (lane == 0) {
        float4 a = { p[0], p[1], p[2], p[3] }, b = { p[4], p[5], p[6], p[7] };
        *(float4*)(probs + (size_t)gw * NE)     = a;
        *(float4*)(probs + (size_t)gw * NE + 4) = b;
        sel_e[2 * gw] = i0; sel_e[2 * gw + 1] = i1;
        sel_w[2 * gw] = v0 * z; sel_w[2 * gw + 1] = v1 * z;
    }
}

// ========= 2) deterministic per-expert list build (1 WG, wave per expert) ==
__global__ void moe_build(const int* __restrict__ sel_e,
                          int* __restrict__ counts, int* __restrict__ offs,
                          int* __restrict__ pair_tok, int* __restrict__ tok_slot) {
    int tid = threadIdx.x, lane = tid & 31, w = tid >> 5;   // w = expert
    for (int sIdx = tid; sIdx < CAP; sIdx += 256) pair_tok[sIdx] = -1;
    __syncthreads();

    __shared__ int s_cnt[NE];
    __shared__ int s_off[NE + 1];

    int cnt = 0;
    for (int base = 0; base < T_TOK; base += 32) {
        int t = base + lane;
        int e0 = sel_e[2 * t], e1 = sel_e[2 * t + 1];
        unsigned m = (unsigned)__ballot((e0 == w) || (e1 == w));
        cnt += __popc(m);
    }
    if (lane == 0) s_cnt[w] = cnt;
    __syncthreads();
    if (tid == 0) {
        int o = 0;
        for (int e = 0; e < NE; ++e) { s_off[e] = o; o += (s_cnt[e] + 63) & ~63; counts[e] = s_cnt[e]; }
        s_off[NE] = o;
        for (int e = 0; e <= NE; ++e) offs[e] = s_off[e];
    }
    __syncthreads();

    int pos = s_off[w];
    for (int base = 0; base < T_TOK; base += 32) {
        int t = base + lane;
        int e0 = sel_e[2 * t], e1 = sel_e[2 * t + 1];
        bool sel = (e0 == w) || (e1 == w);
        unsigned m = (unsigned)__ballot(sel);
        if (sel) {
            int slot = pos + __popc(m & ((1u << lane) - 1u));
            pair_tok[slot] = t;
            int k = (e0 == w) ? 0 : 1;
            tok_slot[2 * t + k] = slot;
        }
        pos += __popc(m);
    }
}

// ================= 3) gather tokens -> bf16 Xg =============================
__global__ void moe_gather(const float* __restrict__ x, const int* __restrict__ pair_tok,
                           unsigned short* __restrict__ xg) {
    size_t idx = (size_t)blockIdx.x * blockDim.x + threadIdx.x;   // CAP*128 threads
    int sIdx = (int)(idx >> 7);
    int c8   = (int)(idx & 127) << 3;
    int t = pair_tok[sIdx];
    uint4 v = { 0u, 0u, 0u, 0u };
    if (t >= 0) {
        const float* xr = x + (size_t)t * DM + c8;
        float4 a = *(const float4*)(xr);
        float4 b = *(const float4*)(xr + 4);
        v.x = pk2(a.x, a.y); v.y = pk2(a.z, a.w);
        v.z = pk2(b.x, b.y); v.w = pk2(b.z, b.w);
    }
    *(uint4*)(xg + (size_t)sIdx * DM + c8) = v;
}

// ================= 4) grouped GEMM via WMMA bf16 ===========================
// C[64 x 128] per WG, 8 waves: wave grid 4(rows) x 2(cols of 64).
// A: bf16 row-major (lda = KDIM), async global->LDS copy (ASYNCcnt).
// B: fp32 weights, row-contiguous b128 loads, packed-transposed into LDS.
#define A_STR 56    // ushort stride: 112B rows, 16B-aligned, 28-dword stride (conflict-free)
#define B_STR 56
template<int KDIM, int NDIM, bool FFN1>
__global__ void moe_gemm(const unsigned short* __restrict__ Abase,
                         const float* __restrict__ Wbase,
                         unsigned short* __restrict__ Hout, float* __restrict__ Yout,
                         const int* __restrict__ offs, const int* __restrict__ counts) {
    __shared__ __align__(16) unsigned short As[2][64 * A_STR];
    __shared__ __align__(16) unsigned short Bs[2][128 * B_STR];

    int row0  = blockIdx.y * 64;
    int total = offs[NE];
    if (row0 >= total) return;
    int e = 0;
    while (offs[e + 1] <= row0) ++e;
    int eEnd = offs[e] + counts[e];
    if (row0 >= eEnd) return;                       // padding-only tile

    const float* W = Wbase + (size_t)e * KDIM * NDIM;
    int colBase = blockIdx.x * 128;

    int tid = threadIdx.x;
    int wid = tid >> 5, lane = tid & 31;
    int wr = wid & 3, wc = wid >> 2;
    int m16 = lane & 15, kh = lane >> 4;

    auto stageA = [&](int kk, int buf) {
        int r = tid >> 2, ch = tid & 3;             // 64 rows x 4 chunks of 16B
        const unsigned short* src = Abase + (size_t)(row0 + r) * KDIM + kk * 32 + ch * 8;
        unsigned short* dst = &As[buf][r * A_STR + ch * 8];
#if USE_ASYNC_LDS
        __builtin_amdgcn_global_load_async_to_lds_b128(
            (gv4i_p)(uintptr_t)src, (sv4i_p)(uintptr_t)dst, 0, 0);
#else
        uint4 v = *(const uint4*)src;
        *(uint4*)dst = v;
#endif
    };
    auto stageB = [&](int kk, int buf) {
        int p  = tid >> 4;                          // 0..15 : k-pair (rows 2p, 2p+1)
        int c0 = (tid & 15) * 8;                    // 8-column group
        const float* src = W + (size_t)(kk * 32 + 2 * p) * NDIM + colBase + c0;
        __builtin_prefetch(src + (size_t)32 * NDIM, 0, 1);
        float4 r0a = *(const float4*)(src);
        float4 r0b = *(const float4*)(src + 4);
        float4 r1a = *(const float4*)(src + NDIM);
        float4 r1b = *(const float4*)(src + NDIM + 4);
        // column c gets packed pair (W[2p][c], W[2p+1][c]) at ushort idx c*B_STR + 2p
        *(unsigned*)&Bs[buf][(c0 + 0) * B_STR + 2 * p] = pk2(r0a.x, r1a.x);
        *(unsigned*)&Bs[buf][(c0 + 1) * B_STR + 2 * p] = pk2(r0a.y, r1a.y);
        *(unsigned*)&Bs[buf][(c0 + 2) * B_STR + 2 * p] = pk2(r0a.z, r1a.z);
        *(unsigned*)&Bs[buf][(c0 + 3) * B_STR + 2 * p] = pk2(r0a.w, r1a.w);
        *(unsigned*)&Bs[buf][(c0 + 4) * B_STR + 2 * p] = pk2(r0b.x, r1b.x);
        *(unsigned*)&Bs[buf][(c0 + 5) * B_STR + 2 * p] = pk2(r0b.y, r1b.y);
        *(unsigned*)&Bs[buf][(c0 + 6) * B_STR + 2 * p] = pk2(r0b.z, r1b.z);
        *(unsigned*)&Bs[buf][(c0 + 7) * B_STR + 2 * p] = pk2(r0b.w, r1b.w);
    };

    v8f acc0 = (v8f)(0.f), acc1 = (v8f)(0.f), acc2 = (v8f)(0.f), acc3 = (v8f)(0.f);

    constexpr int nk = KDIM / 32;
    stageA(0, 0); stageB(0, 0);
    for (int kk = 0; kk < nk; ++kk) {
        wait_async0();
        __syncthreads();
        int buf = kk & 1;
        if (kk + 1 < nk) { stageA(kk + 1, buf ^ 1); stageB(kk + 1, buf ^ 1); }

        // ---- load ALL operands into distinct regs first, then 4 WMMAs ----
        const unsigned short* ap = &As[buf][(wr * 16 + m16) * A_STR];
        uint4 a0 = *(const uint4*)(ap + kh * 8);
        uint4 a1 = *(const uint4*)(ap + 16 + kh * 8);
        const unsigned short* bp = &Bs[buf][(wc * 64 + m16) * B_STR + kh * 8];
        uint4 b00 = *(const uint4*)(bp);
        uint4 b01 = *(const uint4*)(bp + 16);
        uint4 b10 = *(const uint4*)(bp + 16 * B_STR);
        uint4 b11 = *(const uint4*)(bp + 16 * B_STR + 16);
        uint4 b20 = *(const uint4*)(bp + 32 * B_STR);
        uint4 b21 = *(const uint4*)(bp + 32 * B_STR + 16);
        uint4 b30 = *(const uint4*)(bp + 48 * B_STR);
        uint4 b31 = *(const uint4*)(bp + 48 * B_STR + 16);
        v16bf av  = mk16(a0, a1);
        v16bf bv0 = mk16(b00, b01), bv1 = mk16(b10, b11);
        v16bf bv2 = mk16(b20, b21), bv3 = mk16(b30, b31);
        acc0 = __builtin_amdgcn_wmma_f32_16x16x32_bf16(false, av, false, bv0, (short)0, acc0, false, false);
        acc1 = __builtin_amdgcn_wmma_f32_16x16x32_bf16(false, av, false, bv1, (short)0, acc1, false, false);
        acc2 = __builtin_amdgcn_wmma_f32_16x16x32_bf16(false, av, false, bv2, (short)0, acc2, false, false);
        acc3 = __builtin_amdgcn_wmma_f32_16x16x32_bf16(false, av, false, bv3, (short)0, acc3, false, false);
    }

    // epilogue: VGPR r -> M = (kh? 8+r : r), N = lane%16
    int mrow = row0 + wr * 16 + kh * 8;
    v8f accs[4] = { acc0, acc1, acc2, acc3 };
#pragma unroll
    for (int sI = 0; sI < 4; ++sI) {
        int col = colBase + wc * 64 + sI * 16 + m16;
#pragma unroll
        for (int r = 0; r < 8; ++r) {
            int grow = mrow + r;
            if (grow < eEnd) {
                float v = accs[sI][r];
                if (FFN1) {
                    v = 0.5f * v * (1.f + erff(v * 0.70710678118654752f));   // exact GELU
                    Hout[(size_t)grow * NDIM + col] = bf16_of(v);
                } else {
                    Yout[(size_t)grow * NDIM + col] = v;
                }
            }
        }
    }
}

// ================= 5) combine: out = w0*Y[s0] + w1*Y[s1] ===================
__global__ void moe_combine(const float* __restrict__ Y, const int* __restrict__ tok_slot,
                            const float* __restrict__ sel_w, float* __restrict__ out) {
    size_t idx = (size_t)blockIdx.x * blockDim.x + threadIdx.x;  // T*256 threads
    int t  = (int)(idx >> 8);
    int c4 = (int)(idx & 255) << 2;
    int s0 = tok_slot[2 * t], s1 = tok_slot[2 * t + 1];
    float w0 = sel_w[2 * t], w1 = sel_w[2 * t + 1];
    float4 a = *(const float4*)(Y + (size_t)s0 * DM + c4);
    float4 b = *(const float4*)(Y + (size_t)s1 * DM + c4);
    float4 o = { w0 * a.x + w1 * b.x, w0 * a.y + w1 * b.y,
                 w0 * a.z + w1 * b.z, w0 * a.w + w1 * b.w };
    *(float4*)(out + (size_t)t * DM + c4) = o;
}

// ================= 6) load-balance loss ====================================
__global__ void moe_loss(const float* __restrict__ probs, const int* __restrict__ counts,
                         float* __restrict__ out_loss) {
    __shared__ float red[256];
    __shared__ float router[NE];
    float acc[NE];
#pragma unroll
    for (int e = 0; e < NE; ++e) acc[e] = 0.f;
    for (int t = threadIdx.x; t < T_TOK; t += 256)
#pragma unroll
        for (int e = 0; e < NE; ++e) acc[e] += probs[(size_t)t * NE + e];
    for (int e = 0; e < NE; ++e) {
        red[threadIdx.x] = acc[e]; __syncthreads();
        for (int sI = 128; sI > 0; sI >>= 1) {
            if (threadIdx.x < sI) red[threadIdx.x] += red[threadIdx.x + sI];
            __syncthreads();
        }
        if (threadIdx.x == 0) router[e] = red[0] / (float)T_TOK;
        __syncthreads();
    }
    if (threadIdx.x == 0) {
        float l = 0.f, tot = (float)(T_TOK * TOPK);
        for (int e = 0; e < NE; ++e) l += router[e] * ((float)counts[e] / tot);
        out_loss[0] = (float)NE * l;
    }
}

// ================= launcher ================================================
extern "C" void kernel_launch(void* const* d_in, const int* in_sizes, int n_in,
                              void* d_out, int out_size, void* d_ws, size_t ws_size,
                              hipStream_t stream) {
    const float* x  = (const float*)d_in[0];
    const float* Wg = (const float*)d_in[1];
    const float* W1 = (const float*)d_in[2];
    const float* W2 = (const float*)d_in[3];
    float* out = (float*)d_out;                         // T*DM floats + 1 loss
    char*  ws  = (char*)d_ws;

    float*          probs    = (float*)(ws + OFF_PROBS);
    int*            sel_e    = (int*)(ws + OFF_SELE);
    float*          sel_w    = (float*)(ws + OFF_SELW);
    int*            counts   = (int*)(ws + OFF_CNT);
    int*            offs     = (int*)(ws + OFF_OFFS);
    int*            pair_tok = (int*)(ws + OFF_PTOK);
    int*            tok_slot = (int*)(ws + OFF_TSLOT);
    unsigned short* xg       = (unsigned short*)(ws + OFF_XG);
    unsigned short* H        = (unsigned short*)(ws + OFF_H);
    float*          Y        = (float*)(ws + OFF_Y);

    moe_gate<<<T_TOK / 8, 256, 0, stream>>>(x, Wg, probs, sel_e, sel_w);
    moe_build<<<1, 256, 0, stream>>>(sel_e, counts, offs, pair_tok, tok_slot);
    moe_gather<<<(CAP * 128) / 256, 256, 0, stream>>>(x, pair_tok, xg);
    // FFN1: [CAP x 1024] @ [1024 x 4096] -> gelu -> H (bf16)
    moe_gemm<DM, DFF, true><<<dim3(DFF / 128, NROWT), 256, 0, stream>>>(
        xg, W1, H, nullptr, offs, counts);
    // FFN2: [CAP x 4096] @ [4096 x 1024] -> Y (fp32)
    moe_gemm<DFF, DM, false><<<dim3(DM / 128, NROWT), 256, 0, stream>>>(
        H, W2, nullptr, Y, offs, counts);
    moe_combine<<<(T_TOK * 256) / 256, 256, 0, stream>>>(Y, tok_slot, sel_w, out);
    moe_loss<<<1, 256, 0, stream>>>(probs, counts, out + (size_t)T_TOK * DM);
}
